// GINNodeFeatureUpdate_44255343018792
// MI455X (gfx1250) — compile-verified, matrix operand order checked by reading, not run on hardware
//
#include <hip/hip_runtime.h>
#include <hip/hip_bf16.h>
#include <stdint.h>

#define NN 50000
#define NE 800000
#define DIN 128
#define DHID 256

typedef __attribute__((ext_vector_type(16))) __bf16       v16bf;
typedef __attribute__((ext_vector_type(8)))  float        v8f;
typedef __attribute__((ext_vector_type(4)))  unsigned int u32x4;
typedef __attribute__((ext_vector_type(4)))  int          v4i;

#if __has_builtin(__builtin_amdgcn_global_load_async_to_lds_b128)
#define HAVE_ASYNC_LDS 1
typedef __attribute__((address_space(1))) v4i* gv4i_p;  // global int4*
typedef __attribute__((address_space(3))) v4i* lv4i_p;  // LDS int4*
#else
#define HAVE_ASYNC_LDS 0
#endif

// fp32 -> bf16 (RNE-ish)
__device__ __forceinline__ unsigned int pack2bf(float a, float b) {
  unsigned int ua = __float_as_uint(a);
  unsigned int ub = __float_as_uint(b);
  ua += 0x7FFFu + ((ua >> 16) & 1u);
  ub += 0x7FFFu + ((ub >> 16) & 1u);
  return (ua >> 16) | (ub & 0xFFFF0000u);
}
__device__ __forceinline__ unsigned short f2bf(float a) {
  unsigned int ua = __float_as_uint(a);
  ua += 0x7FFFu + ((ua >> 16) & 1u);
  return (unsigned short)(ua >> 16);
}

// -------------------- elementwise: out = (1+eps)*x (float4) --------------------
__global__ void scale_copy_kernel(const float4* __restrict__ x,
                                  const float* __restrict__ eps,
                                  float4* __restrict__ out, int n4) {
  float s = 1.0f + eps[0];
  for (int i = blockIdx.x * blockDim.x + threadIdx.x; i < n4;
       i += gridDim.x * blockDim.x) {
    float4 v = x[i];
    out[i] = make_float4(s * v.x, s * v.y, s * v.z, s * v.w);
  }
}

// -------------------- fp32 -> packed bf16 pairs (4 floats/thread) ------------
__global__ void conv_bf16_kernel(const float4* __restrict__ in,
                                 uint2* __restrict__ out, int nquads) {
  for (int i = blockIdx.x * blockDim.x + threadIdx.x; i < nquads;
       i += gridDim.x * blockDim.x) {
    float4 v = in[i];
    out[i] = make_uint2(pack2bf(v.x, v.y), pack2bf(v.z, v.w));
  }
}

// -------------------- scatter-add: one wave per edge row --------------------
// Indices loaded once per wave; row gathered as coalesced float4s.
template <int LOGD>
__global__ void scatter_add_kernel(const float* __restrict__ x,
                                   const long long* __restrict__ ei,
                                   float* __restrict__ agg) {
  const int D   = 1 << LOGD;
  const int FPL = D >> 5;  // floats per lane (4 or 8)
  const int lane = threadIdx.x & 31;
  const int wid  = (blockIdx.x * blockDim.x + threadIdx.x) >> 5;
  const int nwaves = (gridDim.x * blockDim.x) >> 5;
  for (int e = wid; e < NE; e += nwaves) {
    const long long s = ei[e];       // wave-uniform (single cacheline)
    const long long d = ei[NE + e];
    const float* xs = x + s * D + lane * FPL;
    float*       ad = agg + d * D + lane * FPL;
#pragma unroll
    for (int k = 0; k < FPL; k += 4) {
      float4 v = *(const float4*)(xs + k);
      atomicAdd(ad + k + 0, v.x);
      atomicAdd(ad + k + 1, v.y);
      atomicAdd(ad + k + 2, v.z);
      atomicAdd(ad + k + 3, v.w);
    }
  }
}

// ---- repack fp32 weights [K x Nout] into per-lane bf16 B-fragment order ----
// P dword index: ((kt*ntiles + nt)*32 + lane)*8 + v
__global__ void repack_w_kernel(const float* __restrict__ W,
                                unsigned int* __restrict__ P, int K, int Nout) {
  int ktiles = K >> 5, ntiles = Nout >> 4;
  int total = ktiles * ntiles * 32;
  int t = blockIdx.x * blockDim.x + threadIdx.x;
  if (t >= total) return;
  int lane = t & 31;
  int tile = t >> 5;
  int nt = tile % ntiles;
  int kt = tile / ntiles;
  int n   = (nt << 4) + (lane & 15);
  int hi8 = (lane & 16) ? 8 : 0;
  unsigned int* dst = P + (size_t)t * 8;
#pragma unroll
  for (int v = 0; v < 8; ++v) {
    int k  = (v < 4) ? (hi8 + 2 * v) : (16 + hi8 + 2 * (v - 4));
    int kk = (kt << 5) + k;
    dst[v] = pack2bf(W[(size_t)kk * Nout + n], W[(size_t)(kk + 1) * Nout + n]);
  }
}

// -------------------- WMMA GEMM: C = act(A @ W + b) --------------------
// Block = 8 waves; all waves share one 128-col strip (B staged in LDS once),
// each wave owns one 16-row tile -> 16x128 outputs, 8 f32 accumulators.
// A and B fragments are both explicitly double-buffered.
template <bool OUT_BF16, bool RELU>
__global__ __launch_bounds__(256) void gemm_bias_kernel(
    const unsigned int* __restrict__ Abf,  // [M x K/2] dwords (bf16 pairs)
    const unsigned int* __restrict__ Bp,   // packed bf16 fragments
    const float* __restrict__ bias,        // [Nout]
    void* __restrict__ Cout,               // fp32 [MxNout] or bf16 [MxNout]
    int M, int K, int Nout) {
  const int lane   = threadIdx.x & 31;
  const int wave   = threadIdx.x >> 5;
  const int ktiles = K >> 5;
  const int ntiles = Nout >> 4;
  const int ns     = blockIdx.y;       // 128-col strip id
  const int mtiles = M >> 4;

  __shared__ unsigned int ldsB[16384];  // up to 8 kt * 8 frag * 32 lane * 8 dw

  // ---- stage this strip's B fragments into LDS (contiguous 8KB per kt) ----
  {
    const int tid = threadIdx.x;
    for (int u = tid; u < ktiles * 512; u += 256) {  // 16B units
      const int kt  = u >> 9;
      const int off = u & 511;
      const unsigned int* src =
          Bp + ((size_t)(kt * ntiles + (ns << 3)) << 8) + off * 4;
      unsigned int* dst = &ldsB[(kt << 11) + off * 4];
#if HAVE_ASYNC_LDS
      __builtin_amdgcn_global_load_async_to_lds_b128((gv4i_p)src, (lv4i_p)dst,
                                                     0, 0);
#else
      *(u32x4*)dst = *(const u32x4*)src;
#endif
    }
#if HAVE_ASYNC_LDS
    asm volatile("s_wait_asynccnt 0x0" ::: "memory");
#endif
  }
  __syncthreads();

  int mt = blockIdx.x * 8 + wave;
  const bool valid = (mt < mtiles);
  if (!valid) mt = mtiles - 1;  // clamp: keep wave alive for EXEC=all-1 WMMA

  const int m16 = lane & 15;
  const int hi8 = (lane & 16) ? 8 : 0;
  const unsigned int* arow = Abf + (size_t)((mt << 4) + m16) * (K >> 1);

  union Frag { u32x4 q[2]; v16bf v; };

  v8f acc[8];
#pragma unroll
  for (int j = 0; j < 8; ++j) acc[j] = v8f{};

  // prime A and B double-buffers
  Frag af_cur, af_nxt, bw_cur, bw_nxt;
  {
    const unsigned int* ap = arow + (hi8 >> 1);
    af_cur.q[0] = *(const u32x4*)ap;
    af_cur.q[1] = *(const u32x4*)(ap + 8);
    const unsigned int* lp = &ldsB[lane * 8];
    bw_cur.q[0] = *(const u32x4*)lp;
    bw_cur.q[1] = *(const u32x4*)(lp + 4);
  }

  for (int kt = 0; kt < ktiles; ++kt) {
    // prefetch next A fragment (redundant reload of last on final iter)
    {
      int ktn = (kt + 1 < ktiles) ? (kt + 1) : kt;
      const unsigned int* ap = arow + (ktn << 4) + (hi8 >> 1);
      af_nxt.q[0] = *(const u32x4*)ap;
      af_nxt.q[1] = *(const u32x4*)(ap + 8);
    }
#pragma unroll
    for (int j = 0; j < 8; ++j) {
      // prefetch next B fragment: (kt, j+1), or (kt+1, 0), clamped at end
      int ktn = (j == 7) ? kt + 1 : kt;
      int jn  = (j == 7) ? 0 : j + 1;
      if (ktn >= ktiles) { ktn = kt; jn = j; }
      const unsigned int* lp = &ldsB[(ktn << 11) + (jn << 8) + lane * 8];
      bw_nxt.q[0] = *(const u32x4*)lp;
      bw_nxt.q[1] = *(const u32x4*)(lp + 4);

      acc[j] = __builtin_amdgcn_wmma_f32_16x16x32_bf16(
          false, af_cur.v, false, bw_cur.v, (short)0, acc[j], false, false);
      bw_cur = bw_nxt;
    }
    af_cur = af_nxt;
  }

  if (valid) {
#pragma unroll
    for (int j = 0; j < 8; ++j) {
      const int col = (ns << 7) + (j << 4) + m16;
      const float bv = bias[col];
#pragma unroll
      for (int r = 0; r < 8; ++r) {
        const int row = (mt << 4) + hi8 + r;
        float v = acc[j][r] + bv;
        if (RELU) v = fmaxf(v, 0.0f);
        if constexpr (OUT_BF16)
          ((unsigned short*)Cout)[(size_t)row * Nout + col] = f2bf(v);
        else
          ((float*)Cout)[(size_t)row * Nout + col] = v;
      }
    }
  }
}

// -------------------- launch --------------------
extern "C" void kernel_launch(void* const* d_in, const int* in_sizes, int n_in,
                              void* d_out, int out_size, void* d_ws,
                              size_t ws_size, hipStream_t stream) {
  const float*     x    = (const float*)d_in[0];
  const long long* ei   = (const long long*)d_in[1];
  const float*     eps0 = (const float*)d_in[2];
  const float*     W1_0 = (const float*)d_in[3];
  const float*     b1_0 = (const float*)d_in[4];
  const float*     W2_0 = (const float*)d_in[5];
  const float*     b2_0 = (const float*)d_in[6];
  const float*     eps1 = (const float*)d_in[7];
  const float*     W1_1 = (const float*)d_in[8];
  const float*     b1_1 = (const float*)d_in[9];
  const float*     W2_1 = (const float*)d_in[10];
  const float*     b2_1 = (const float*)d_in[11];
  const float*     Wo   = (const float*)d_in[12];
  const float*     bo   = (const float*)d_in[13];
  float* out = (float*)d_out;

  // workspace carve-up (all chunks 256B-aligned by construction)
  char* ws = (char*)d_ws;
  float*        agg  = (float*)ws;        ws += (size_t)NN * DHID * 4;  // fp32 accum
  float*        h2   = (float*)ws;        ws += (size_t)NN * DHID * 4;  // fp32 layer-0 out
  unsigned int* abf  = (unsigned int*)ws; ws += (size_t)NN * DHID * 2;  // bf16 act (reused)
  unsigned int* h1bf = (unsigned int*)ws; ws += (size_t)NN * DHID * 2;  // bf16 hidden
  unsigned int* pW1_0 = (unsigned int*)ws; ws += (size_t)DIN  * DHID * 2;
  unsigned int* pW2_0 = (unsigned int*)ws; ws += (size_t)DHID * DHID * 2;
  unsigned int* pW1_1 = (unsigned int*)ws; ws += (size_t)DHID * DHID * 2;
  unsigned int* pW2_1 = (unsigned int*)ws; ws += (size_t)DHID * DHID * 2;
  unsigned int* pWo   = (unsigned int*)ws; ws += (size_t)DHID * DIN  * 2;

  auto repack = [&](const float* W, unsigned int* P, int K, int Nout) {
    int total = (K >> 5) * (Nout >> 4) * 32;
    repack_w_kernel<<<(total + 255) / 256, 256, 0, stream>>>(W, P, K, Nout);
  };
  repack(W1_0, pW1_0, DIN,  DHID);
  repack(W2_0, pW2_0, DHID, DHID);
  repack(W1_1, pW1_1, DHID, DHID);
  repack(W2_1, pW2_1, DHID, DHID);
  repack(Wo,   pWo,   DHID, DIN);

  const int mtiles = NN >> 4;               // 3125
  const dim3 blk(256);

  // ---- layer 0 (128 -> 256) ----
  {
    int n = NN * DIN;
    scale_copy_kernel<<<(n / 4 + 255) / 256, blk, 0, stream>>>(
        (const float4*)x, eps0, (float4*)agg, n / 4);
    scatter_add_kernel<7><<<12500, blk, 0, stream>>>(x, ei, agg);
    conv_bf16_kernel<<<(n / 4 + 255) / 256, blk, 0, stream>>>(
        (const float4*)agg, (uint2*)abf, n / 4);
    dim3 g((mtiles + 7) / 8, DHID >> 7);
    gemm_bias_kernel<true, true><<<g, blk, 0, stream>>>(
        abf, pW1_0, b1_0, h1bf, NN, DIN, DHID);
    gemm_bias_kernel<false, false><<<g, blk, 0, stream>>>(
        h1bf, pW2_0, b2_0, h2, NN, DHID, DHID);
  }
  // ---- layer 1 (256 -> 256) ----
  {
    int n = NN * DHID;
    scale_copy_kernel<<<(n / 4 + 255) / 256, blk, 0, stream>>>(
        (const float4*)h2, eps1, (float4*)agg, n / 4);
    scatter_add_kernel<8><<<12500, blk, 0, stream>>>(h2, ei, agg);
    conv_bf16_kernel<<<(n / 4 + 255) / 256, blk, 0, stream>>>(
        (const float4*)agg, (uint2*)abf, n / 4);
    dim3 g((mtiles + 7) / 8, DHID >> 7);
    gemm_bias_kernel<true, true><<<g, blk, 0, stream>>>(
        abf, pW1_1, b1_1, h1bf, NN, DHID, DHID);
    // layer-1 output feeds only fc_out -> write bf16 directly (reuse abf)
    gemm_bias_kernel<true, false><<<g, blk, 0, stream>>>(
        h1bf, pW2_1, b2_1, abf, NN, DHID, DHID);
  }
  // ---- fc_out (256 -> 128) ----
  {
    dim3 g((mtiles + 7) / 8, DIN >> 7);
    gemm_bias_kernel<false, false><<<g, blk, 0, stream>>>(
        abf, pWo, bo, out, NN, DHID, DIN);
  }
}
